// TransformerEncoderLayer_18494129176578
// MI455X (gfx1250) — matrix operand order repaired and checked by reading, not compiled, not run on hardware
//
#include <hip/hip_runtime.h>
#include <stdint.h>

// ---------------------------------------------------------------------------
// Transformer encoder layer (B=2,S=2048,E=768,H=12,D=64,FF=3072) for gfx1250.
// All GEMMs via v_wmma_f32_16x16x32_bf16 (wave32, 16x16 tiles, f32 accum).
// Double-buffered LDS staging; non-temporal hints on the 400MB attn stream.
// ---------------------------------------------------------------------------

#define BB 2
#define SS 2048
#define EE 768
#define HH 12
#define DD 64
#define FFF 3072

typedef __attribute__((ext_vector_type(16))) __bf16 v16bf;
typedef __attribute__((ext_vector_type(8)))  float  v8f;
typedef __attribute__((ext_vector_type(4)))  float  f32x4;   // native clang vector (NT-load capable)

union Frag16 { v16bf v; unsigned int u[8]; };

__device__ __forceinline__ unsigned short f32_to_bf16(float f) {
    unsigned int x = __float_as_uint(f);
    unsigned int r = x + 0x7FFFu + ((x >> 16) & 1u);   // round-to-nearest-even
    return (unsigned short)(r >> 16);
}
__device__ __forceinline__ unsigned int pack_bf16x2(float a, float b) {
    return (unsigned int)f32_to_bf16(a) | ((unsigned int)f32_to_bf16(b) << 16);
}

// ---------------------------------------------------------------------------
// elementwise f32 -> bf16
// ---------------------------------------------------------------------------
__global__ void __launch_bounds__(256) cvt_f32_to_bf16(const float* __restrict__ in,
                                                       unsigned short* __restrict__ out,
                                                       int n) {
    int i = blockIdx.x * 256 + threadIdx.x;
    if (i < n) out[i] = f32_to_bf16(in[i]);
}

// ---------------------------------------------------------------------------
// Generic bf16 GEMM:  C[M,N] = A[M,K] @ B[K,N] + bias, row-major, optional ReLU,
// output f32 or bf16.  Block tile 128x128x32, 256 threads (8 waves),
// wave tile 32(M) x 64(N) -> 2x4 accumulators.  Double-buffered LDS.
// ---------------------------------------------------------------------------
template <bool RELU, bool OUT_BF16>
__global__ void __launch_bounds__(256) gemm_bf16_kernel(
    const unsigned short* __restrict__ A,
    const unsigned short* __restrict__ Bm,
    const float* __restrict__ bias,
    float* __restrict__ Cf,
    unsigned short* __restrict__ Cb,
    int M, int N, int K)
{
    __shared__ unsigned short As[2][128 * 32];   // [m][k]
    __shared__ unsigned short Bs[2][128 * 32];   // [n][k]  (transposed on store)

    const int tid  = threadIdx.x;
    const int wave = tid >> 5;
    const int lane = tid & 31;
    const int lh   = lane >> 4;     // lane half (0/1)
    const int ln   = lane & 15;
    const int wm   = wave >> 1;     // 0..3  (32-row wave tiles)
    const int wn   = wave & 1;      // 0..1  (64-col wave tiles)
    const int m0   = blockIdx.y * 128;
    const int n0   = blockIdx.x * 128;

    const int lm   = tid >> 1;      // 0..127 : A load row
    const int lseg = tid & 1;       // 0..1   : A load k-half (16 each)
    const int kp   = tid >> 4;      // 0..15  : B load k-pair (rows 2kp,2kp+1)
    const int ns   = tid & 15;      // 0..15  : B load n-segment (8 each)

    uint4 aLo, aHi, bLo, bHi;
    auto load_tiles = [&](int k0) {
        const unsigned short* Ag = A + (size_t)(m0 + lm) * K + k0 + lseg * 16;
        aLo = *(const uint4*)(Ag);
        aHi = *(const uint4*)(Ag + 8);
        const unsigned short* Bg = Bm + (size_t)(k0 + 2 * kp) * N + n0 + ns * 8;
        bLo = *(const uint4*)(Bg);
        bHi = *(const uint4*)(Bg + N);
    };
    auto store_tiles = [&](int buf) {
        *(uint4*)&As[buf][lm * 32 + lseg * 16]     = aLo;
        *(uint4*)&As[buf][lm * 32 + lseg * 16 + 8] = aHi;
        unsigned short lo[8], hi[8];
        *(uint4*)lo = bLo;
        *(uint4*)hi = bHi;
        unsigned int* base = (unsigned int*)&Bs[buf][0];
        #pragma unroll
        for (int j = 0; j < 8; ++j)   // Bs[n][k], packed k-pair store
            base[((ns * 8 + j) * 32 + 2 * kp) >> 1] =
                (unsigned int)lo[j] | ((unsigned int)hi[j] << 16);
    };

    v8f acc[2][4] = {};

    const int nT = K / 32;
    load_tiles(0);
    store_tiles(0);
    __syncthreads();

    for (int t = 0; t < nT; ++t) {
        const int buf = t & 1;
        if (t + 1 < nT) load_tiles((t + 1) * 32);

        Frag16 afr[2], bfr[4];
        #pragma unroll
        for (int mi = 0; mi < 2; ++mi) {
            const int m = wm * 32 + mi * 16 + ln;
            #pragma unroll
            for (int j = 0; j < 8; ++j) {
                const int kb = 2 * j + lh * 16;
                afr[mi].u[j] = *(const unsigned int*)&As[buf][m * 32 + kb];
            }
        }
        #pragma unroll
        for (int ni = 0; ni < 4; ++ni) {
            const int n = wn * 64 + ni * 16 + ln;
            #pragma unroll
            for (int j = 0; j < 8; ++j) {
                const int kb = 2 * j + lh * 16;
                bfr[ni].u[j] = *(const unsigned int*)&Bs[buf][n * 32 + kb];
            }
        }
        #pragma unroll
        for (int mi = 0; mi < 2; ++mi)
            #pragma unroll
            for (int ni = 0; ni < 4; ++ni)
                acc[mi][ni] = __builtin_amdgcn_wmma_f32_16x16x32_bf16(
                    false, afr[mi].v, false, bfr[ni].v,
                    (short)0, acc[mi][ni], false, false);

        if (t + 1 < nT) store_tiles(buf ^ 1);
        __syncthreads();
    }

    // ---- epilogue ----
    #pragma unroll
    for (int mi = 0; mi < 2; ++mi) {
        #pragma unroll
        for (int ni = 0; ni < 4; ++ni) {
            const int n  = n0 + wn * 64 + ni * 16 + ln;
            const float bv = bias ? bias[n] : 0.f;
            #pragma unroll
            for (int r = 0; r < 8; ++r) {
                const int m = m0 + wm * 32 + mi * 16 + lh * 8 + r;
                float v = acc[mi][ni][r] + bv;
                if (RELU) v = fmaxf(v, 0.f);
                if (OUT_BF16) Cb[(size_t)m * N + n] = f32_to_bf16(v);
                else          Cf[(size_t)m * N + n] = v;
            }
        }
    }
}

// ---------------------------------------------------------------------------
// Attention scores: attn[b,h,q,k] = (q.k)/sqrt(D) - |q-k|   (pre-softmax)
// grid = (S/128, S/128, B*H), block 256.  Single LDS stage (K=64 only).
// ---------------------------------------------------------------------------
__global__ void __launch_bounds__(256) attn_scores_kernel(
    const unsigned short* __restrict__ qb,
    const unsigned short* __restrict__ kb,
    float* __restrict__ attn)
{
    __shared__ unsigned short Qs[128 * 64];   // [m][d]
    __shared__ unsigned short Ks[128 * 64];   // [n][d]

    const int tid = threadIdx.x;
    const int bh  = blockIdx.z;
    const int b   = bh / HH;
    const int h   = bh % HH;
    const int m0  = blockIdx.y * 128;
    const int n0  = blockIdx.x * 128;

    const int lm   = tid >> 1;
    const int lseg = tid & 1;    // d half (32 each)
    {
        const size_t qoff = (size_t)(b * SS + m0 + lm) * EE + h * DD + lseg * 32;
        const uint4* s = (const uint4*)(qb + qoff);
        uint4* d = (uint4*)&Qs[lm * 64 + lseg * 32];
        d[0] = s[0]; d[1] = s[1]; d[2] = s[2]; d[3] = s[3];
        const size_t koff = (size_t)(b * SS + n0 + lm) * EE + h * DD + lseg * 32;
        const uint4* s2 = (const uint4*)(kb + koff);
        uint4* d2 = (uint4*)&Ks[lm * 64 + lseg * 32];
        d2[0] = s2[0]; d2[1] = s2[1]; d2[2] = s2[2]; d2[3] = s2[3];
    }
    __syncthreads();

    const int wave = tid >> 5, lane = tid & 31, lh = lane >> 4, ln = lane & 15;
    const int wm = wave >> 1, wn = wave & 1;   // 32(M) x 64(N) wave tile

    v8f acc[2][4] = {};
    #pragma unroll
    for (int d0 = 0; d0 < DD; d0 += 32) {
        Frag16 afr[2], bfr[4];
        #pragma unroll
        for (int mi = 0; mi < 2; ++mi) {
            const int m = wm * 32 + mi * 16 + ln;
            #pragma unroll
            for (int j = 0; j < 8; ++j) {
                const int kbase = 2 * j + lh * 16;
                afr[mi].u[j] = *(const unsigned int*)&Qs[m * 64 + d0 + kbase];
            }
        }
        #pragma unroll
        for (int ni = 0; ni < 4; ++ni) {
            const int n = wn * 64 + ni * 16 + ln;
            #pragma unroll
            for (int j = 0; j < 8; ++j) {
                const int kbase = 2 * j + lh * 16;
                bfr[ni].u[j] = *(const unsigned int*)&Ks[n * 64 + d0 + kbase];
            }
        }
        #pragma unroll
        for (int mi = 0; mi < 2; ++mi)
            #pragma unroll
            for (int ni = 0; ni < 4; ++ni)
                acc[mi][ni] = __builtin_amdgcn_wmma_f32_16x16x32_bf16(
                    false, afr[mi].v, false, bfr[ni].v,
                    (short)0, acc[mi][ni], false, false);
    }

    const float scale = 0.125f;   // 1/sqrt(64)
    #pragma unroll
    for (int mi = 0; mi < 2; ++mi) {
        #pragma unroll
        for (int ni = 0; ni < 4; ++ni) {
            const int n = n0 + wn * 64 + ni * 16 + ln;
            #pragma unroll
            for (int r = 0; r < 8; ++r) {
                const int m = m0 + wm * 32 + mi * 16 + lh * 8 + r;
                const float alibi = -fabsf((float)(m - n));   // SCALING = 1.0
                __builtin_nontemporal_store(acc[mi][ni][r] * scale + alibi,
                                            &attn[((size_t)bh * SS + m) * SS + n]);
            }
        }
    }
}

// ---------------------------------------------------------------------------
// Row softmax in place, one row of S=2048 per block (256 thr x 8 elems)
// ---------------------------------------------------------------------------
__global__ void __launch_bounds__(256) softmax_kernel(float* __restrict__ attn)
{
    __shared__ float red[256];
    float* p = attn + (size_t)blockIdx.x * SS;
    const int t = threadIdx.x;

    float v[8];
    float mx = -3.4e38f;
    #pragma unroll
    for (int i = 0; i < 8; ++i) {
        v[i] = __builtin_nontemporal_load(&p[t + i * 256]);
        mx = fmaxf(mx, v[i]);
    }
    red[t] = mx; __syncthreads();
    for (int s = 128; s > 0; s >>= 1) { if (t < s) red[t] = fmaxf(red[t], red[t + s]); __syncthreads(); }
    mx = red[0]; __syncthreads();

    float sum = 0.f;
    #pragma unroll
    for (int i = 0; i < 8; ++i) { v[i] = __expf(v[i] - mx); sum += v[i]; }
    red[t] = sum; __syncthreads();
    for (int s = 128; s > 0; s >>= 1) { if (t < s) red[t] += red[t + s]; __syncthreads(); }
    const float inv = 1.f / red[0];
    #pragma unroll
    for (int i = 0; i < 8; ++i)
        __builtin_nontemporal_store(v[i] * inv, &p[t + i * 256]);
}

// ---------------------------------------------------------------------------
// ctx[b,s,h,:] = attn[b,h,s,:] @ v[b,:,h,:]  -> bf16 out
// Per (b,h): GEMM M=S, N=64, K=S.  grid = (S/128, B*H), block 256.
// Wave tile 32x32 -> 2x2 accumulators.  attn read NT f32, packed to bf16.
// Double-buffered LDS.
// ---------------------------------------------------------------------------
__global__ void __launch_bounds__(256) attn_ctx_kernel(
    const float* __restrict__ attn,
    const unsigned short* __restrict__ vb,
    unsigned short* __restrict__ ctxb)
{
    __shared__ unsigned short As[2][128 * 32];   // [m][k]
    __shared__ unsigned short Vs[2][64 * 32];    // [n][k]

    const int tid = threadIdx.x;
    const int bh  = blockIdx.y;
    const int b   = bh / HH;
    const int h   = bh % HH;
    const int m0  = blockIdx.x * 128;

    const int wave = tid >> 5, lane = tid & 31, lh = lane >> 4, ln = lane & 15;
    const int wm = wave >> 1, wn = wave & 1;    // 32(M) x 32(N) wave tile

    const int lm   = tid >> 1;   // attn row
    const int lseg = tid & 1;    // 16 floats each
    const int kp   = tid >> 4;   // 0..15 : v k-pair (rows 2kp,2kp+1)
    const int ns   = tid & 15;   // 0..15 : v n-segment (4 each)

    f32x4 f0, f1, f2, f3;
    uint2 vLo, vHi;
    auto load_tiles = [&](int k0) {
        const float* Ag = attn + ((size_t)bh * SS + m0 + lm) * SS + k0 + lseg * 16;
        f0 = __builtin_nontemporal_load((const f32x4*)(Ag + 0));
        f1 = __builtin_nontemporal_load((const f32x4*)(Ag + 4));
        f2 = __builtin_nontemporal_load((const f32x4*)(Ag + 8));
        f3 = __builtin_nontemporal_load((const f32x4*)(Ag + 12));
        const unsigned short* Vg = vb + (size_t)(b * SS + k0 + 2 * kp) * EE + h * DD + ns * 4;
        vLo = *(const uint2*)(Vg);
        vHi = *(const uint2*)(Vg + EE);
    };
    auto store_tiles = [&](int buf) {
        unsigned int* dst = (unsigned int*)&As[buf][lm * 32 + lseg * 16];
        dst[0] = pack_bf16x2(f0.x, f0.y);
        dst[1] = pack_bf16x2(f0.z, f0.w);
        dst[2] = pack_bf16x2(f1.x, f1.y);
        dst[3] = pack_bf16x2(f1.z, f1.w);
        dst[4] = pack_bf16x2(f2.x, f2.y);
        dst[5] = pack_bf16x2(f2.z, f2.w);
        dst[6] = pack_bf16x2(f3.x, f3.y);
        dst[7] = pack_bf16x2(f3.z, f3.w);
        unsigned short lo[4], hi[4];
        *(uint2*)lo = vLo;
        *(uint2*)hi = vHi;
        unsigned int* base = (unsigned int*)&Vs[buf][0];
        #pragma unroll
        for (int j = 0; j < 4; ++j)   // Vs[n][k], packed k-pair store
            base[((ns * 4 + j) * 32 + 2 * kp) >> 1] =
                (unsigned int)lo[j] | ((unsigned int)hi[j] << 16);
    };

    v8f acc[2][2] = {};

    const int nT = SS / 32;
    load_tiles(0);
    store_tiles(0);
    __syncthreads();

    for (int t = 0; t < nT; ++t) {
        const int buf = t & 1;
        if (t + 1 < nT) load_tiles((t + 1) * 32);

        Frag16 afr[2], bfr[2];
        #pragma unroll
        for (int mi = 0; mi < 2; ++mi) {
            const int m = wm * 32 + mi * 16 + ln;
            #pragma unroll
            for (int j = 0; j < 8; ++j) {
                const int kb = 2 * j + lh * 16;
                afr[mi].u[j] = *(const unsigned int*)&As[buf][m * 32 + kb];
            }
        }
        #pragma unroll
        for (int ni = 0; ni < 2; ++ni) {
            const int n = wn * 32 + ni * 16 + ln;
            #pragma unroll
            for (int j = 0; j < 8; ++j) {
                const int kb = 2 * j + lh * 16;
                bfr[ni].u[j] = *(const unsigned int*)&Vs[buf][n * 32 + kb];
            }
        }
        #pragma unroll
        for (int mi = 0; mi < 2; ++mi)
            #pragma unroll
            for (int ni = 0; ni < 2; ++ni)
                acc[mi][ni] = __builtin_amdgcn_wmma_f32_16x16x32_bf16(
                    false, afr[mi].v, false, bfr[ni].v,
                    (short)0, acc[mi][ni], false, false);

        if (t + 1 < nT) store_tiles(buf ^ 1);
        __syncthreads();
    }

    #pragma unroll
    for (int mi = 0; mi < 2; ++mi) {
        #pragma unroll
        for (int ni = 0; ni < 2; ++ni) {
            const int e = h * DD + wn * 32 + ni * 16 + ln;
            #pragma unroll
            for (int r = 0; r < 8; ++r) {
                const int m = m0 + wm * 32 + mi * 16 + lh * 8 + r;
                ctxb[(size_t)(b * SS + m) * EE + e] = f32_to_bf16(acc[mi][ni][r]);
            }
        }
    }
}

// ---------------------------------------------------------------------------
// LayerNorm with fused residual: out = LN(xa + xr)*g + beta
// one row (E=768) per block; optional f32 and/or bf16 outputs
// ---------------------------------------------------------------------------
__global__ void __launch_bounds__(256) ln_kernel(
    const float* __restrict__ xa, const float* __restrict__ xr,
    const float* __restrict__ g,  const float* __restrict__ beta,
    float* __restrict__ outf, unsigned short* __restrict__ outb)
{
    __shared__ float r1[256], r2[256];
    const size_t row = blockIdx.x;
    const int t = threadIdx.x;
    const float* pa = xa + row * EE;
    const float* pb = xr + row * EE;

    float v[3]; float s = 0.f, ss = 0.f;
    #pragma unroll
    for (int i = 0; i < 3; ++i) {
        v[i] = pa[t + i * 256] + pb[t + i * 256];
        s += v[i]; ss += v[i] * v[i];
    }
    r1[t] = s; r2[t] = ss; __syncthreads();
    for (int k = 128; k > 0; k >>= 1) {
        if (t < k) { r1[t] += r1[t + k]; r2[t] += r2[t + k]; }
        __syncthreads();
    }
    const float mean = r1[0] * (1.f / EE);
    const float var  = r2[0] * (1.f / EE) - mean * mean;
    const float rs   = rsqrtf(var + 1e-5f);
    #pragma unroll
    for (int i = 0; i < 3; ++i) {
        const int c = t + i * 256;
        const float o = (v[i] - mean) * rs * g[c] + beta[c];
        if (outf) outf[row * EE + c] = o;
        if (outb) outb[row * EE + c] = f32_to_bf16(o);
    }
}

// ---------------------------------------------------------------------------
// Host-side orchestration
// ---------------------------------------------------------------------------
extern "C" void kernel_launch(void* const* d_in, const int* in_sizes, int n_in,
                              void* d_out, int out_size, void* d_ws, size_t ws_size,
                              hipStream_t stream) {
    const float* x   = (const float*)d_in[0];
    const float* Wq  = (const float*)d_in[1];
    const float* bq  = (const float*)d_in[2];
    const float* Wk  = (const float*)d_in[3];
    const float* bk  = (const float*)d_in[4];
    const float* Wv  = (const float*)d_in[5];
    const float* bv  = (const float*)d_in[6];
    const float* Wo  = (const float*)d_in[7];
    const float* bo  = (const float*)d_in[8];
    const float* W1  = (const float*)d_in[9];
    const float* b1  = (const float*)d_in[10];
    const float* W2  = (const float*)d_in[11];
    const float* b2  = (const float*)d_in[12];
    const float* g1  = (const float*)d_in[13];
    const float* be1 = (const float*)d_in[14];
    const float* g2  = (const float*)d_in[15];
    const float* be2 = (const float*)d_in[16];

    float* out  = (float*)d_out;                      // [B,S,E]
    float* attn = out + (size_t)BB * SS * EE;         // [B,H,S,S]

    // workspace sub-allocation (256B aligned)
    char* ws = (char*)d_ws;
    size_t off = 0;
    auto alloc = [&](size_t bytes) -> void* {
        void* p = ws + off;
        off = (off + bytes + 255) & ~(size_t)255;
        return p;
    };
    const size_t MSE = (size_t)BB * SS * EE;          // 3,145,728
    unsigned short* xb   = (unsigned short*)alloc(MSE * 2);
    unsigned short* wqb  = (unsigned short*)alloc((size_t)EE * EE * 2);
    unsigned short* wkb  = (unsigned short*)alloc((size_t)EE * EE * 2);
    unsigned short* wvb  = (unsigned short*)alloc((size_t)EE * EE * 2);
    unsigned short* wob  = (unsigned short*)alloc((size_t)EE * EE * 2);
    unsigned short* w1b  = (unsigned short*)alloc((size_t)EE * FFF * 2);
    unsigned short* w2b  = (unsigned short*)alloc((size_t)FFF * EE * 2);
    unsigned short* qbuf = (unsigned short*)alloc(MSE * 2);
    unsigned short* kbuf = (unsigned short*)alloc(MSE * 2);
    unsigned short* vbuf = (unsigned short*)alloc(MSE * 2);
    unsigned short* ctxb = (unsigned short*)alloc(MSE * 2);
    float*          aout = (float*)alloc(MSE * 4);    // attention out (f32)
    float*          hbuf = (float*)alloc(MSE * 4);    // LN1 result (f32)
    unsigned short* hb   = (unsigned short*)alloc(MSE * 2);
    unsigned short* ff1b = (unsigned short*)alloc((size_t)BB * SS * FFF * 2);
    float*          ff2  = (float*)alloc(MSE * 4);

    const int M = BB * SS;                            // 4096

    auto cvt = [&](const float* src, unsigned short* dst, size_t n) {
        cvt_f32_to_bf16<<<dim3((unsigned)((n + 255) / 256)), 256, 0, stream>>>(src, dst, (int)n);
    };

    // 1) bf16 conversions
    cvt(x,  xb,  MSE);
    cvt(Wq, wqb, (size_t)EE * EE);
    cvt(Wk, wkb, (size_t)EE * EE);
    cvt(Wv, wvb, (size_t)EE * EE);
    cvt(Wo, wob, (size_t)EE * EE);
    cvt(W1, w1b, (size_t)EE * FFF);
    cvt(W2, w2b, (size_t)FFF * EE);

    // 2) Q/K/V projections (bf16 out)
    dim3 gqkv(EE / 128, M / 128);
    gemm_bf16_kernel<false, true><<<gqkv, 256, 0, stream>>>(xb, wqb, bq, nullptr, qbuf, M, EE, EE);
    gemm_bf16_kernel<false, true><<<gqkv, 256, 0, stream>>>(xb, wkb, bk, nullptr, kbuf, M, EE, EE);
    gemm_bf16_kernel<false, true><<<gqkv, 256, 0, stream>>>(xb, wvb, bv, nullptr, vbuf, M, EE, EE);

    // 3) scores + ALiBi, then row softmax in place (attn lives in d_out)
    attn_scores_kernel<<<dim3(SS / 128, SS / 128, BB * HH), 256, 0, stream>>>(qbuf, kbuf, attn);
    softmax_kernel<<<dim3(BB * HH * SS), 256, 0, stream>>>(attn);

    // 4) ctx = attn @ v  (bf16 out)
    attn_ctx_kernel<<<dim3(SS / 128, BB * HH), 256, 0, stream>>>(attn, vbuf, ctxb);

    // 5) output projection (f32 out)
    gemm_bf16_kernel<false, false><<<dim3(EE / 128, M / 128), 256, 0, stream>>>(
        ctxb, wob, bo, aout, nullptr, M, EE, EE);

    // 6) h = LN(x + attn_out)  -> f32 + bf16
    ln_kernel<<<dim3(M), 256, 0, stream>>>(x, aout, g1, be1, hbuf, hb);

    // 7) FFN: relu(h @ W1 + b1) -> bf16 ; @ W2 + b2 -> f32
    gemm_bf16_kernel<true, true><<<dim3(FFF / 128, M / 128), 256, 0, stream>>>(
        hb, w1b, b1, nullptr, ff1b, M, FFF, EE);
    gemm_bf16_kernel<false, false><<<dim3(EE / 128, M / 128), 256, 0, stream>>>(
        ff1b, w2b, b2, ff2, nullptr, M, EE, FFF);

    // 8) out = LN(h + ff)
    ln_kernel<<<dim3(M), 256, 0, stream>>>(hbuf, ff2, g2, be2, out, nullptr);
}
